// Head_82471962018242
// MI455X (gfx1250) — matrix operand order repaired and checked by reading, not compiled, not run on hardware
//
#include <hip/hip_runtime.h>
#include <hip/hip_bf16.h>

typedef __attribute__((ext_vector_type(16))) _Float16 v16h;
typedef __attribute__((ext_vector_type(8)))  _Float16 v8h;
typedef __attribute__((ext_vector_type(8)))  float    v8f;
typedef __attribute__((ext_vector_type(4)))  unsigned int u32x4;
typedef __attribute__((ext_vector_type(8)))  int          i32x8;
typedef __attribute__((ext_vector_type(4)))  int          i32x4;

constexpr int B = 8;
constexpr int T = 2048;
constexpr int C = 1024;
constexpr int H = 128;
constexpr float SCALE = 0.08838834764831845f; // 128^-0.5

// LDS layout (single __shared__ block at LDS offset 0), f16 elements:
//   K^T bufs: [2][128 h][32 keys] elem off buf*4096,        byte buf*8192
//   V bufs  : [2][32 keys][128 h] elem off 8192 + buf*4096, byte 16384 + buf*8192
//   P stage : [8 waves][16][32]   elem off 16384 + wave*512, byte 32768 + wave*1024
constexpr int LDS_ELEMS = 2 * 128 * 32 + 2 * 32 * 128 + 8 * 16 * 32; // 20480 (40KB)

__device__ __forceinline__ v8f wmma16x16x32(v16h a, v16h b, v8f c) {
    return __builtin_amdgcn_wmma_f32_16x16x32_f16(false, a, false, b,
                                                  (short)0, c, false, false);
}

// Build a v16h A/B operand from two contiguous 16-byte halves.
__device__ __forceinline__ v16h cat8h(v8h lo, v8h hi) {
    return __builtin_shufflevector(lo, hi, 0, 1, 2, 3, 4, 5, 6, 7,
                                   8, 9, 10, 11, 12, 13, 14, 15);
}

// TDM 2D tile load into LDS. D# per CDNA5 ISA ch.8:
// group0 = {count=1, lds_addr, global_addr, type=2}; group1 = dims/strides.
// This toolchain exposes the 6-arg builtin: (g0, g1, g2, g3, g4, cpol).
__device__ __forceinline__ void tdm_load_tile(unsigned lds_byte_off,
                                              const _Float16* gsrc,
                                              int tile_d0, int tile_d1,
                                              int tensor_d0, int tensor_d1,
                                              int stride0) {
    unsigned long long ga = (unsigned long long)gsrc;
    u32x4 g0;
    g0[0] = 1u;                                           // count=1, user mode
    g0[1] = lds_byte_off;                                 // lds_addr [63:32]
    g0[2] = (unsigned)(ga & 0xffffffffull);               // global_addr lo
    g0[3] = (unsigned)((ga >> 32) & 0x1ffffffull)         // global_addr hi
            | (2u << 30);                                 // type=2 ("image")
    i32x8 g1;
    g1[0] = (1 << 16);                                    // data_size=1 (2B)
    g1[1] = (tensor_d0 & 0xffff) << 16;                   // tensor_dim0 lo16
    g1[2] = (int)(((unsigned)tensor_d0 >> 16) |
                  (((unsigned)tensor_d1 & 0xffffu) << 16)); // d0 hi | d1 lo16
    g1[3] = (int)(((unsigned)tensor_d1 >> 16) |
                  (((unsigned)tile_d0 & 0xffffu) << 16)); // d1 hi | tile_dim0
    g1[4] = tile_d1 & 0xffff;                             // tile_dim1|tile_dim2=0
    g1[5] = stride0;                                      // tensor_dim0_stride lo
    g1[6] = 0;
    g1[7] = 0;
    i32x4 z4 = {0, 0, 0, 0};                              // 2D: groups 2/3 unused
    i32x8 z8 = {0, 0, 0, 0, 0, 0, 0, 0};
    __builtin_amdgcn_tensor_load_to_lds(g0, g1, z4, z4, z8, 0);
}

// ---------------- Stage 1: q/k/v projections (f32 x, f32 W -> f16) ----------
// q, v stored [B,T,H]; k stored TRANSPOSED [B,H,T] so stage-2 B-operands are
// contiguous in the key dimension.
__global__ __launch_bounds__(32) void proj_kernel(
    const float* __restrict__ x,
    const float* __restrict__ Wq, const float* __restrict__ Wk,
    const float* __restrict__ Wv,
    _Float16* __restrict__ qh, _Float16* __restrict__ khT,
    _Float16* __restrict__ vh) {
    const int lane = threadIdx.x & 31;
    const int tile = blockIdx.x;              // B*(T/16) tiles
    const int bb   = tile / (T / 16);
    const int t0   = (tile % (T / 16)) * 16;
    const int h0   = blockIdx.y * 16;
    const int which = blockIdx.z;             // 0=q 1=k 2=v

    const float* Wm = (which == 0) ? Wq : (which == 1) ? Wk : Wv;
    const float oscale = (which == 0) ? SCALE : 1.0f;

    const int row = lane & 15;
    const int kb  = (lane < 16) ? 0 : 8;

    v8f acc = {};
    const float* xrow = x + (size_t)(bb * T + t0 + row) * C;
    for (int c0 = 0; c0 < C; c0 += 32) {
        // A operand: rows of x, two contiguous 8-float chunks per lane
        const v8f xa0 = *(const v8f*)(xrow + c0 + kb);
        const v8f xa1 = *(const v8f*)(xrow + c0 + 16 + kb);
        // B operand: column h0+(lane&15) of W, k in same order as A
        const float* wcol = Wm + (size_t)c0 * H + h0 + (lane & 15);
        v16h a, w;
#pragma unroll
        for (int i = 0; i < 8; ++i) {
            a[i]     = (_Float16)xa0[i];
            a[8 + i] = (_Float16)xa1[i];
            w[i]     = (_Float16)wcol[(size_t)(kb + i) * H];
            w[8 + i] = (_Float16)wcol[(size_t)(16 + kb + i) * H];
        }
        acc = wmma16x16x32(a, w, acc);
    }

    const int half = lane >> 4;
    const int col  = lane & 15;
    v8h vals;
#pragma unroll
    for (int r = 0; r < 8; ++r) vals[r] = (_Float16)(acc[r] * oscale);

    if (which == 1) {
        // k transposed [B,H,T]: per lane the 8 values are contiguous in T
        // (rows t0+8*half+0..7, fixed column h0+col) -> one b128 store.
        *(v8h*)(khT + (size_t)(bb * H + h0 + col) * T + t0 + 8 * half) = vals;
    } else {
        _Float16* outp = (which == 0) ? qh : vh;
#pragma unroll
        for (int r = 0; r < 8; ++r)
            outp[(size_t)(bb * T + t0 + r + 8 * half) * H + h0 + col] = vals[r];
    }
}

// ------- Stage 2: causal flash attention, 8 waves / 128 queries per block ---
// Wave 0 drives double-buffered TDM loads of K^T / V chunks into LDS; all
// waves consume them with v_wmma_f32_16x16x32_f16 + online softmax.
__global__ __launch_bounds__(256) void attn_kernel(
    const _Float16* __restrict__ qh, const _Float16* __restrict__ khT,
    const _Float16* __restrict__ vh, float* __restrict__ out) {
    __shared__ __align__(128) _Float16 lds[LDS_ELEMS];

    const int lane = threadIdx.x & 31;
    const int wave = threadIdx.x >> 5;        // 0..7
    const int bb   = blockIdx.x / (T / 128);
    const int q0b  = (blockIdx.x % (T / 128)) * 128;  // block's first query row
    const int qw0  = q0b + wave * 16;                 // wave's first query row
    const int row  = lane & 15;
    const int half = lane >> 4;
    const int kb   = half * 8;
    const int colN = lane & 15;

    _Float16* Pw = lds + 16384 + wave * 512;  // per-wave P staging [16][32]

    // Q tile in A layout: two contiguous v8h halves per lane per 32-h chunk
    v16h qa[4];
    const _Float16* qrow = qh + (size_t)(bb * T + qw0 + row) * H;
#pragma unroll
    for (int hc = 0; hc < 4; ++hc)
        qa[hc] = cat8h(*(const v8h*)(qrow + hc * 32 + kb),
                       *(const v8h*)(qrow + hc * 32 + 16 + kb));

    float m[8], l[8];
    v8f o[8] = {};
#pragma unroll
    for (int r = 0; r < 8; ++r) { m[r] = -INFINITY; l[r] = 0.0f; }

    const int nkblk = (q0b + 127) / 32 + 1;   // key chunks the whole block needs

    // Prologue: wave 0 kicks off chunk 0 into buffer 0.
    if (wave == 0) {
        // K^T: tensor [B*H, T] stride T; tile 128 h-rows x 32 keys
        tdm_load_tile(0u, khT + (size_t)bb * H * T, 32, 128, T, B * H, T);
        // V:   tensor [B*T, H] stride H; tile 32 key-rows x 128 h
        tdm_load_tile(16384u, vh + (size_t)bb * T * H, H, 32, H, B * T, H);
    }

    for (int kt = 0; kt < nkblk; ++kt) {
        const int k0  = kt * 32;
        const int buf = kt & 1;

        if (wave == 0) __builtin_amdgcn_s_wait_tensorcnt(0);
        __syncthreads();                       // chunk kt resident in LDS[buf]

        if (wave == 0 && kt + 1 < nkblk) {     // prefetch kt+1 into other buffer
            const int nbuf = 1 - buf;
            tdm_load_tile((unsigned)(nbuf * 8192),
                          khT + (size_t)bb * H * T + (k0 + 32),
                          32, 128, T, B * H, T);
            tdm_load_tile((unsigned)(16384 + nbuf * 8192),
                          vh + (size_t)(bb * T + k0 + 32) * H,
                          H, 32, H, B * T, H);
        }

        if (k0 <= qw0 + 15) {                  // wave-uniform causal early-out
            const _Float16* KbT = lds + buf * 4096;         // [128 h][32 keys]
            const _Float16* Vb  = lds + 8192 + buf * 4096;  // [32 keys][128 h]

            // S = Q * K^T (two 16x16 tiles over 32 keys)
            v8f s0 = {}, s1 = {};
#pragma unroll
            for (int hc = 0; hc < 4; ++hc) {
                // B layout: column = key (lane&15); elements over h in A's k order
                const _Float16* kcol = KbT + (size_t)(hc * 32) * 32 + colN;
                v16h bk0, bk1;
#pragma unroll
                for (int i = 0; i < 8; ++i) {
                    bk0[i]     = kcol[(kb + i) * 32];
                    bk0[8 + i] = kcol[(16 + kb + i) * 32];
                    bk1[i]     = kcol[(kb + i) * 32 + 16];
                    bk1[8 + i] = kcol[(16 + kb + i) * 32 + 16];
                }
                s0 = wmma16x16x32(qa[hc], bk0, s0);
                s1 = wmma16x16x32(qa[hc], bk1, s1);
            }

            // causal mask + online softmax (C layout: row=r+8*half, col=colN)
#pragma unroll
            for (int r = 0; r < 8; ++r) {
                const int qg = qw0 + r + 8 * half;
                if (k0 + colN > qg)      s0[r] = -INFINITY;
                if (k0 + 16 + colN > qg) s1[r] = -INFINITY;
            }
#pragma unroll
            for (int r = 0; r < 8; ++r) {
                float mx = fmaxf(s0[r], s1[r]);
#pragma unroll
                for (int d = 1; d < 16; d <<= 1)
                    mx = fmaxf(mx, __shfl_xor(mx, d, 32));
                const float mnew  = fmaxf(m[r], mx);
                const float alpha = __expf(m[r] - mnew);
                const float p0 = __expf(s0[r] - mnew);
                const float p1 = __expf(s1[r] - mnew);
                float sum = p0 + p1;
#pragma unroll
                for (int d = 1; d < 16; d <<= 1)
                    sum += __shfl_xor(sum, d, 32);
                l[r] = l[r] * alpha + sum;
                m[r] = mnew;
#pragma unroll
                for (int ht = 0; ht < 8; ++ht) o[ht][r] *= alpha;
                const int rowM = r + 8 * half;
                Pw[rowM * 32 + colN]      = (_Float16)p0;
                Pw[rowM * 32 + colN + 16] = (_Float16)p1;
            }
            // Same-wave DS ops are in-order (ISA 7.3); no barrier needed here.

            // P in A layout: two contiguous v8h halves per lane
            const v16h pa = cat8h(*(const v8h*)(Pw + row * 32 + kb),
                                  *(const v8h*)(Pw + row * 32 + 16 + kb));

            // O += P * V over 8 h-tiles (B layout: column = h; elements over keys in A's k order)
#pragma unroll
            for (int ht = 0; ht < 8; ++ht) {
                v16h vb;
#pragma unroll
                for (int i = 0; i < 8; ++i) {
                    vb[i]     = Vb[(kb + i) * 128 + ht * 16 + colN];
                    vb[8 + i] = Vb[(16 + kb + i) * 128 + ht * 16 + colN];
                }
                o[ht] = wmma16x16x32(pa, vb, o[ht]);
            }
        }
        __syncthreads();                       // all reads of LDS[buf] done
    }

    // normalize and store fp32 output
#pragma unroll
    for (int r = 0; r < 8; ++r) {
        const int rowM = r + 8 * half;
        const float inv = 1.0f / l[r];
#pragma unroll
        for (int ht = 0; ht < 8; ++ht)
            out[(size_t)(bb * T + qw0 + rowM) * H + ht * 16 + colN] =
                o[ht][r] * inv;
    }
}

extern "C" void kernel_launch(void* const* d_in, const int* in_sizes, int n_in,
                              void* d_out, int out_size, void* d_ws, size_t ws_size,
                              hipStream_t stream) {
    const float* x  = (const float*)d_in[0];
    const float* Wk = (const float*)d_in[1];
    const float* Wq = (const float*)d_in[2];
    const float* Wv = (const float*)d_in[3];
    float* out = (float*)d_out;

    const size_t BTH = (size_t)B * T * H;  // 2,097,152 elements
    _Float16* qh  = (_Float16*)d_ws;
    _Float16* khT = qh + BTH;              // [B, H, T] transposed
    _Float16* vh  = khT + BTH;             // 12 MB total f16 scratch

    dim3 pgrid(B * (T / 16), H / 16, 3);
    proj_kernel<<<pgrid, 32, 0, stream>>>(x, Wq, Wk, Wv, qh, khT, vh);

    dim3 agrid(B * (T / 128));
    attn_kernel<<<agrid, 256, 0, stream>>>(qh, khT, vh, out);
}
